// SO3Convolution_36163624632815
// MI455X (gfx1250) — matrix operand order repaired and checked by reading, not compiled, not run on hardware
//
#include <hip/hip_runtime.h>

typedef __attribute__((ext_vector_type(16))) _Float16 v16h;
typedef __attribute__((ext_vector_type(8)))  _Float16 v8h;
typedef __attribute__((ext_vector_type(8)))  float    v8f;
typedef __attribute__((vector_size(16)))     int      vi4;   // int4 as expected by async builtin

#define N_NODES_C 50000
#define N_EDGES_C 800000
#define F_IN_C   16
#define F_OUT_C  16
#define N_RBF_C  16
#define HID_C    64
#define NH_C     9
#define CUTOFF_C 5.0f

// gfx1250 async global->LDS copy (ASYNCcnt-tracked). Builtin name is
// toolchain-dependent; fall back to synchronous cooperative copies if absent.
#if defined(__has_builtin)
#if __has_builtin(__builtin_amdgcn_global_load_async_to_lds_b128)
#define HAVE_ASYNC_LDS 1
#endif
#endif

#if defined(HAVE_ASYNC_LDS)
// Compile-time unrolled async row copy: IOFFSET must be an ICE, so bake the
// per-chunk byte offset (J*16) into each call via template recursion. The ISA
// adds INST_OFFSET to both the global and the LDS address.
template <int J, int N>
struct AsyncRowCopy {
  static __device__ __forceinline__ void run(float* gp, float* lp) {
    __builtin_amdgcn_global_load_async_to_lds_b128(
        (__attribute__((address_space(1))) vi4*)gp,
        (__attribute__((address_space(3))) vi4*)lp,
        J * 16, 0);
    AsyncRowCopy<J + 1, N>::run(gp, lp);
  }
};
template <int N>
struct AsyncRowCopy<N, N> {
  static __device__ __forceinline__ void run(float*, float*) {}
};
#endif

// Fast SiLU: one v_exp_f32 + one v_rcp_f32 (avoids the v_div_scale IEEE chain).
__device__ __forceinline__ float silu_f(float v) {
  return v * __builtin_amdgcn_rcpf(1.0f + __expf(-v));
}

// Load a 16x32 f16 A-fragment from an LDS tile stored row-major [16][row_stride].
// Contiguous K order (matches packed B): lane L holds M=L&15;
//   halves 0..15 -> K = 16*(L>>4) + 0..15
__device__ __forceinline__ v16h load_a_frag(const _Float16* s, int row_stride, int koff, int lane) {
  const int M    = lane & 15;
  const int base = (lane >> 4) * 16;
  const _Float16* p = s + M * row_stride + koff + base;
  v8h lo = *(const v8h*)(p);
  v8h hi = *(const v8h*)(p + 8);
  v16h a;
#pragma unroll
  for (int i = 0; i < 8; ++i) { a[i] = lo[i]; a[i + 8] = hi[i]; }
  return a;
}

__global__ void zero_out_kernel(float4* __restrict__ out, int n4) {
  int i = blockIdx.x * blockDim.x + threadIdx.x;
  if (i < n4) out[i] = make_float4(0.f, 0.f, 0.f, 0.f);
}

// Pre-swizzle W1/W2/W3 into per-lane B-fragment order (f16).
// B fragment (32K x 16N tile): element h of lane L is  W[k][n],
//   k = ktile*32 + 16*(L>>4) + h,  n = ntile*16 + (L&15).
// Fragment idx layout: frag*512 + L*16 + h.
__global__ void pack_weights_kernel(const float* __restrict__ W1,
                                    const float* __restrict__ W2,
                                    const float* __restrict__ W3,
                                    _Float16* __restrict__ ws) {
  _Float16* W1B = ws;                     // 4 frags  (N=64,  K=32 incl. zero pad)
  _Float16* W2B = ws + 4 * 512;           // 8 frags  (N=64,  K=64)
  _Float16* W3B = ws + 12 * 512;          // 32 frags (N=256, K=64)
  const int tid = blockIdx.x * blockDim.x + threadIdx.x;
  const int stride = gridDim.x * blockDim.x;

  for (int idx = tid; idx < 4 * 512; idx += stride) {
    int t = idx >> 9, L = (idx >> 4) & 31, h = idx & 15;
    int k = ((L >> 4) * 16) + h, n = t * 16 + (L & 15);
    W1B[idx] = (k < N_RBF_C) ? (_Float16)W1[k * HID_C + n] : (_Float16)0.0f;
  }
  for (int idx = tid; idx < 8 * 512; idx += stride) {
    int frag = idx >> 9, L = (idx >> 4) & 31, h = idx & 15;
    int t = frag >> 1, kt = frag & 1;
    int k = kt * 32 + ((L >> 4) * 16) + h, n = t * 16 + (L & 15);
    W2B[idx] = (_Float16)W2[k * HID_C + n];
  }
  for (int idx = tid; idx < 32 * 512; idx += stride) {
    int frag = idx >> 9, L = (idx >> 4) & 31, h = idx & 15;
    int t = frag >> 1, kt = frag & 1;
    int k = kt * 32 + ((L >> 4) * 16) + h, n = t * 16 + (L & 15);
    W3B[idx] = (_Float16)W3[k * (F_IN_C * F_OUT_C) + n];
  }
}

// One wave (32 lanes) processes a tile of 16 edges end-to-end.
__global__ void __launch_bounds__(32)
so3_fused_kernel(const float* __restrict__ x,
                 const int*   __restrict__ ei,
                 const float* __restrict__ ev,
                 const float* __restrict__ b1,
                 const float* __restrict__ b2,
                 const float* __restrict__ b3,
                 const float* __restrict__ widths,
                 const _Float16* __restrict__ W1B,
                 const _Float16* __restrict__ W2B,
                 const _Float16* __restrict__ W3B,
                 float* __restrict__ out) {
  __shared__ _Float16 sA [16 * 32];    // RBF A-tile (K padded to 32)
  __shared__ _Float16 sH1[16 * 64];    // layer-1 activations
  __shared__ _Float16 sH2[16 * 64];    // layer-2 activations
  __shared__ _Float16 sRW[16 * 256];   // per-edge weight matrices rw[e][o][i]
  __shared__ __align__(16) float sX[16 * 144];  // gathered x[src[e]] rows
  __shared__ float sSH[16 * 9];
  __shared__ float sCut[16];
  __shared__ int   sSrc[16];
  __shared__ int   sDst[16];

  const int lane  = threadIdx.x;
  const int col   = lane & 15;
  const int rbase = (lane >> 4) * 8;   // C/D rows: M = rbase + r

  // ---------------- Phase 0: per-edge geometry, SH, cutoff, RBF ----------------
  if (lane < 16) {
    const long eg = (long)blockIdx.x * 16 + lane;
    const float vx = ev[eg * 3 + 0], vy = ev[eg * 3 + 1], vz = ev[eg * 3 + 2];
    const float d   = __builtin_amdgcn_sqrtf(vx * vx + vy * vy + vz * vz);
    const float inv = __builtin_amdgcn_rcpf(d + 1e-12f);
    const float nx = vx * inv, ny = vy * inv, nz = vz * inv;
    const float c0 = 0.28209479177387814f, c1 = 0.4886025119029199f,
                c2 = 1.0925484305920792f,  c20 = 0.31539156525252005f,
                c22 = 0.5462742152960396f;
    float sh[9];
    sh[0] = c0;           sh[1] = c1 * ny;     sh[2] = c1 * nz;   sh[3] = c1 * nx;
    sh[4] = c2 * nx * ny; sh[5] = c2 * ny * nz;
    sh[6] = c20 * (3.0f * nz * nz - 1.0f);
    sh[7] = c2 * nx * nz; sh[8] = c22 * (nx * nx - ny * ny);
#pragma unroll
    for (int j = 0; j < 9; ++j) sSH[lane * 9 + j] = sh[j];
    sCut[lane] = (d < CUTOFF_C)
                   ? 0.5f * (1.0f + __cosf(3.14159265358979323f * d / CUTOFF_C))
                   : 0.0f;
    sSrc[lane] = ei[eg];
    sDst[lane] = ei[(long)N_EDGES_C + eg];
#pragma unroll
    for (int k = 0; k < N_RBF_C; ++k) {
      const float c = CUTOFF_C * (float)k / (float)(N_RBF_C - 1);
      const float w = widths[k];
      const float t = d - c;
      sA[lane * 32 + k] = (_Float16)__expf(-(t * t) * __builtin_amdgcn_rcpf(2.0f * w * w));
    }
#pragma unroll
    for (int k = N_RBF_C; k < 32; ++k) sA[lane * 32 + k] = (_Float16)0.0f;
  }
  asm volatile("s_wait_dscnt 0" ::: "memory");

  // ---------------- Gather x[src[e]] rows into LDS (overlaps the whole MLP) ---
  // 2 lanes per edge: lane L copies half (L&1) of row src[L>>1] = 288 B as
  // 18 async b128 chunks that differ only in the IOFFSET immediate.
  {
    const int e    = lane >> 1;
    const int half = lane & 1;
    float* gp = (float*)(x + (long)sSrc[e] * (F_IN_C * NH_C) + half * 72);
    float* lp = &sX[e * 144 + half * 72];
#if defined(HAVE_ASYNC_LDS)
    AsyncRowCopy<0, 18>::run(gp, lp);
#else
#pragma unroll
    for (int j = 0; j < 18; ++j)
      *(float4*)(lp + j * 4) = *(const float4*)(gp + j * 4);
#endif
  }

  // ---------------- Layer 1: (16x16 rbf) x (16x64 W1) ----------------
  {
    const v16h a = load_a_frag(sA, 32, 0, lane);
#pragma unroll
    for (int t = 0; t < 4; ++t) {
      const v16h bf = *(const v16h*)(W1B + ((t * 32 + lane) << 4));
      v8f c = {};
      c = __builtin_amdgcn_wmma_f32_16x16x32_f16(false, a, false, bf, (short)0, c, false, false);
      const float bias = b1[t * 16 + col];
#pragma unroll
      for (int r = 0; r < 8; ++r)
        sH1[(rbase + r) * 64 + t * 16 + col] = (_Float16)silu_f(c[r] + bias);
    }
  }
  asm volatile("s_wait_dscnt 0" ::: "memory");

  // ---------------- Layer 2: (16x64) x (64x64 W2) ----------------
  {
    const v16h a0 = load_a_frag(sH1, 64, 0, lane);
    const v16h a1 = load_a_frag(sH1, 64, 32, lane);
#pragma unroll
    for (int t = 0; t < 4; ++t) {
      const v16h bf0 = *(const v16h*)(W2B + (((t * 2 + 0) * 32 + lane) << 4));
      const v16h bf1 = *(const v16h*)(W2B + (((t * 2 + 1) * 32 + lane) << 4));
      v8f c = {};
      c = __builtin_amdgcn_wmma_f32_16x16x32_f16(false, a0, false, bf0, (short)0, c, false, false);
      c = __builtin_amdgcn_wmma_f32_16x16x32_f16(false, a1, false, bf1, (short)0, c, false, false);
      const float bias = b2[t * 16 + col];
#pragma unroll
      for (int r = 0; r < 8; ++r)
        sH2[(rbase + r) * 64 + t * 16 + col] = (_Float16)silu_f(c[r] + bias);
    }
  }
  asm volatile("s_wait_dscnt 0" ::: "memory");

  // ---------------- Layer 3: (16x64) x (64x256 W3), scale by cutoff ----------------
  {
    const v16h a0 = load_a_frag(sH2, 64, 0, lane);
    const v16h a1 = load_a_frag(sH2, 64, 32, lane);
#pragma unroll 4
    for (int t = 0; t < 16; ++t) {
      const v16h bf0 = *(const v16h*)(W3B + (((t * 2 + 0) * 32 + lane) << 4));
      const v16h bf1 = *(const v16h*)(W3B + (((t * 2 + 1) * 32 + lane) << 4));
      v8f c = {};
      c = __builtin_amdgcn_wmma_f32_16x16x32_f16(false, a0, false, bf0, (short)0, c, false, false);
      c = __builtin_amdgcn_wmma_f32_16x16x32_f16(false, a1, false, bf1, (short)0, c, false, false);
      const float bias = b3[t * 16 + col];
#pragma unroll
      for (int r = 0; r < 8; ++r) {
        const int e = rbase + r;
        sRW[e * 256 + t * 16 + col] = (_Float16)((c[r] + bias) * sCut[e]);
      }
    }
  }
  asm volatile("s_wait_dscnt 0" ::: "memory");
#if defined(HAVE_ASYNC_LDS)
  asm volatile("s_wait_asynccnt 0" ::: "memory");
#endif

  // ---------------- Tensor product + scatter: one WMMA per edge ----------------
#pragma unroll 2
  for (int e = 0; e < 16; ++e) {
    // A = rw_e (M=o, K=i; i>=16 padded with zero).
    v16h arw = {};
    if (rbase == 0) {
      const v8h lo = *(const v8h*)(sRW + e * 256 + col * 16);
      const v8h hi = *(const v8h*)(sRW + e * 256 + col * 16 + 8);
#pragma unroll
      for (int i = 0; i < 8; ++i) { arw[i] = lo[i]; arw[i + 8] = hi[i]; }
      // lanes 16..31 map to K=16..31 -> stay zero
    }
    // B = x[src] (K=i, N=h) from the LDS-staged row. Lanes 0..8 carry valid
    // columns; rest are pad. sh[h] scaling is applied after the WMMA.
    v16h bx = {};
    if (lane < 9) {
#pragma unroll
      for (int i = 0; i < F_IN_C; ++i)
        bx[i] = (_Float16)sX[e * 144 + i * NH_C + lane];
    }
    v8f m = {};
    m = __builtin_amdgcn_wmma_f32_16x16x32_f16(false, arw, false, bx, (short)0, m, false, false);
    if (col < 9) {
      const float shv = sSH[e * 9 + col];
      float* op = out + (long)sDst[e] * (F_OUT_C * NH_C) + col;
#pragma unroll
      for (int r = 0; r < 8; ++r)
        atomicAdd(op + (rbase + r) * NH_C, m[r] * shv);
    }
  }
}

extern "C" void kernel_launch(void* const* d_in, const int* in_sizes, int n_in,
                              void* d_out, int out_size, void* d_ws, size_t ws_size,
                              hipStream_t stream) {
  const float* x      = (const float*)d_in[0];
  const int*   ei     = (const int*)d_in[1];
  const float* ev     = (const float*)d_in[2];
  const float* W1     = (const float*)d_in[3];
  const float* b1     = (const float*)d_in[4];
  const float* W2     = (const float*)d_in[5];
  const float* b2     = (const float*)d_in[6];
  const float* W3     = (const float*)d_in[7];
  const float* b3     = (const float*)d_in[8];
  const float* widths = (const float*)d_in[9];
  float* out = (float*)d_out;
  _Float16* wsf = (_Float16*)d_ws;

  // out is accumulated with atomics -> zero it every launch (graph-replay safe).
  const int n4 = out_size / 4;  // 7,200,000 is divisible by 4
  zero_out_kernel<<<(n4 + 255) / 256, 256, 0, stream>>>((float4*)out, n4);
  // Pack f32 weights into f16 B-fragment order (44 KB, L2-resident afterwards).
  pack_weights_kernel<<<88, 256, 0, stream>>>(W1, W2, W3, wsf);
  // 1 wave per 16-edge tile.
  so3_fused_kernel<<<N_EDGES_C / 16, 32, 0, stream>>>(
      x, ei, ev, b1, b2, b3, widths,
      wsf, wsf + 4 * 512, wsf + 12 * 512, out);
}
